// BandSplitModule_14843406975594
// MI455X (gfx1250) — matrix operand order repaired and checked by reading, not compiled, not run on hardware
//
#include <hip/hip_runtime.h>

typedef __attribute__((ext_vector_type(16))) _Float16 v16h;
typedef __attribute__((ext_vector_type(8)))  _Float16 v8h;
typedef __attribute__((ext_vector_type(4)))  _Float16 v4h;
typedef __attribute__((ext_vector_type(8)))  float    v8f;
typedef __attribute__((ext_vector_type(4)))  float    v4f;

#define NBANDS 38
#define EMB    128
#define NT     512
#define NF     1025
#define LN_EPS 1e-5f

struct BandPtrs {
    const float* g[NBANDS];   // ln_g
    const float* be[NBANDS];  // ln_b
    const float* W[NBANDS];   // (128, feat)
    const float* bi[NBANDS];  // (128,)
};

// One band class per template instantiation: LBW = log2(bandwidth).
// Classes: LBW=4 bands 0-19 (fs 0+32i... actually fs = i*16), LBW=5 bands 20-31,
// LBW=6 bands 32-36, LBW=0 band 37.
template <int LBW>
__global__ __launch_bounds__(256)
void bandsplit_wmma(const float* __restrict__ x,
                    float* __restrict__ out,
                    BandPtrs P, int bandBase, int fsBase)
{
    constexpr int BW   = 1 << LBW;
    constexpr int FEAT = 4 * BW;              // 64 / 128 / 256 / 4
    constexpr int KT   = (FEAT + 31) / 32;    // 2 / 4 / 8 / 1
    constexpr int KP   = KT * 32;
    constexpr int AH_S = KP + 8;              // Ah row stride (halves): 72/136/264/40
                                              // -> dword strides 36/68/132/20, all == 4 mod 64*? (gcd 4) conflict-free

    const int band = bandBase + blockIdx.y;
    const int fs   = fsBase + blockIdx.y * BW;

    const int mtile = blockIdx.x;             // 256 tiles of 16 rows over (b=8) x (t=512)
    const int b  = mtile >> 5;                // 32 tiles per batch
    const int t0 = (mtile & 31) << 4;

    const int tid  = threadIdx.x;
    const int lane = tid & 31;
    const int wv   = tid >> 5;                // wave 0..7 -> N tile

    __shared__ float    As[FEAT][16];         // transposed raw block: [k][t-row]
    __shared__ _Float16 Ah[16][AH_S];         // normalized f16, padded to KP
    __shared__ _Float16 Ws[EMB][40];          // W chunk 128 x 32 (stride 40 halves = 20 dwords)
    __shared__ float    pS[16][17], pQ[16][17];
    __shared__ float    mu[16], rsg[16];

    const float* __restrict__ gp = P.g[band];
    const float* __restrict__ bp = P.be[band];
    const float* __restrict__ Wp = P.W[band];
    const float* __restrict__ ip = P.bi[band];

    // ---- 1) Load 16 x FEAT block as float4 along t (global_load_b128, coalesced).
    //         feature k = ch*BW + df -> x[b, ch, fs+df, t0 + 4*q .. +3]
    constexpr int NV4 = FEAT * 4;             // float4 transactions
    #pragma unroll
    for (int idx = tid; idx < NV4; idx += 256) {
        const int k  = idx >> 2;
        const int rq = (idx & 3) << 2;
        const int ch = k >> LBW;
        const int df = k & (BW - 1);
        const v4f v = *(const v4f*)&x[(((size_t)b * 4 + ch) * NF + (fs + df)) * NT + t0 + rq];
        *(v4f*)&As[k][rq] = v;                // ds_store_b128
    }
    __syncthreads();

    // ---- 2) Per-row mean / var: 16 partials per row, then 16-thread reduce.
    {
        const int r = tid & 15;
        const int j = tid >> 4;
        float s = 0.f, q = 0.f;
        #pragma unroll
        for (int k = j; k < FEAT; k += 16) {
            const float v = As[k][r];
            s += v; q += v * v;
        }
        pS[j][r] = s; pQ[j][r] = q;
    }
    __syncthreads();
    if (tid < 16) {
        float s = 0.f, q = 0.f;
        #pragma unroll
        for (int j = 0; j < 16; ++j) { s += pS[j][tid]; q += pQ[j][tid]; }
        const float m   = s * (1.0f / (float)FEAT);
        const float var = q * (1.0f / (float)FEAT) - m * m;
        mu[tid]  = m;
        rsg[tid] = rsqrtf(var + LN_EPS);
    }
    __syncthreads();

    // ---- 3) Normalize -> f16 into Ah (zero-pad K to KP; only FEAT=4 class pads).
    #pragma unroll
    for (int idx = tid; idx < 16 * KP; idx += 256) {
        const int r = idx & 15;
        const int k = idx >> 4;
        float v = 0.f;
        if (k < FEAT)
            v = (As[k][r] - mu[r]) * rsg[r] * gp[k] + bp[k];
        Ah[r][k] = (_Float16)v;
    }

    // ---- 4) GEMM: acc(16x16) += A(16x32) x B(32x16) over KT k-tiles (unrolled).
    const int n0  = wv << 4;
    const int row = lane & 15;
    const int hi  = (lane >> 4) & 1;

    v8f acc = {};
    #pragma unroll
    for (int kb = 0; kb < KT; ++kb) {
        __syncthreads();                      // Ws reuse / Ah visibility on first iter
        if constexpr (FEAT >= 32) {
            // Stage W chunk: 128 rows x 32 k, 1024 float4 loads -> packed f16 ds_store_b64.
            #pragma unroll
            for (int it = 0; it < 4; ++it) {
                const int idx = tid + it * 256;
                const int n   = idx >> 3;
                const int kk  = (idx & 7) << 2;
                const v4f w = *(const v4f*)&Wp[(size_t)n * FEAT + kb * 32 + kk];
                v4h h;
                h[0] = (_Float16)w[0]; h[1] = (_Float16)w[1];
                h[2] = (_Float16)w[2]; h[3] = (_Float16)w[3];
                *(v4h*)&Ws[n][kk] = h;
            }
        } else {
            // FEAT == 4, KT == 1: one row (4 floats) per thread, zero-pad k=4..31.
            if (tid < EMB) {
                const v4f w = *(const v4f*)&Wp[tid * 4];
                v4h h;
                h[0] = (_Float16)w[0]; h[1] = (_Float16)w[1];
                h[2] = (_Float16)w[2]; h[3] = (_Float16)w[3];
                *(v4h*)&Ws[tid][0] = h;
                const v4h z = {};
                #pragma unroll
                for (int kk = 4; kk < 32; kk += 4)
                    *(v4h*)&Ws[tid][kk] = z;
            }
        }
        __syncthreads();

        // A fragment (ISA layout): lanes 0-15 K {0..7,16..23}; lanes 16-31 K {8..15,24..31}
        v16h a;
        {
            const v8h a0 = *(const v8h*)&Ah[row][kb * 32 + hi * 8];
            const v8h a1 = *(const v8h*)&Ah[row][kb * 32 + 16 + hi * 8];
            #pragma unroll
            for (int e = 0; e < 8; ++e) { a[e] = a0[e]; a[8 + e] = a1[e]; }
        }
        // B fragment: col n0+row; lanes 0-15 K {0..7,16..23}; lanes 16-31 K {8..15,24..31}
        v16h bf;
        {
            const v8h b0 = *(const v8h*)&Ws[n0 + row][hi * 8];
            const v8h b1 = *(const v8h*)&Ws[n0 + row][16 + hi * 8];
            #pragma unroll
            for (int e = 0; e < 8; ++e) { bf[e] = b0[e]; bf[8 + e] = b1[e]; }
        }
        acc = __builtin_amdgcn_wmma_f32_16x16x32_f16(
            false, a, false, bf, (short)0, acc, false, false);
    }

    // ---- 5) Epilogue: D layout (VGPR v -> M = v + 8*hi, lane&15 -> N), add bias.
    const int col = n0 + row;
    const float bv = ip[col];
    const int r0 = hi ? 8 : 0;
    const size_t obase = (((size_t)b * NBANDS + band) * NT + t0) * EMB + col;
    #pragma unroll
    for (int v = 0; v < 8; ++v)
        out[obase + (size_t)(v + r0) * EMB] = acc[v] + bv;
}

extern "C" void kernel_launch(void* const* d_in, const int* in_sizes, int n_in,
                              void* d_out, int out_size, void* d_ws, size_t ws_size,
                              hipStream_t stream) {
    (void)in_sizes; (void)n_in; (void)d_ws; (void)ws_size; (void)out_size;

    // Input order (setup_inputs insertion order): x, ln_g[0..37], ln_b[0..37],
    // W[0..37], b[0..37]  -> 153 pointers.
    const float* x = (const float*)d_in[0];
    float* out = (float*)d_out;
    BandPtrs P;
    for (int i = 0; i < NBANDS; ++i) {
        P.g[i]  = (const float*)d_in[1 + i];
        P.be[i] = (const float*)d_in[1 + NBANDS + i];
        P.W[i]  = (const float*)d_in[1 + 2 * NBANDS + i];
        P.bi[i] = (const float*)d_in[1 + 3 * NBANDS + i];
    }

    const dim3 blk(256);
    // 256 M-tiles x (#bands in class); fs = fsBase + blockIdx.y * BW
    bandsplit_wmma<4><<<dim3(256, 20), blk, 0, stream>>>(x, out, P,  0,    0);
    bandsplit_wmma<5><<<dim3(256, 12), blk, 0, stream>>>(x, out, P, 20,  320);
    bandsplit_wmma<6><<<dim3(256,  5), blk, 0, stream>>>(x, out, P, 32,  704);
    bandsplit_wmma<0><<<dim3(256,  1), blk, 0, stream>>>(x, out, P, 37, 1024);
}
